// graph_extractor_4346506904256
// MI455X (gfx1250) — matrix operand order repaired and checked by reading, not compiled, not run on hardware
//
#include <hip/hip_runtime.h>
#include <hip/hip_bf16.h>

#define NEG_SLOPE 0.2f
#define BN_EPS 1e-5f
#define FLT_NEG_HUGE -3.402823466e38f

typedef __attribute__((ext_vector_type(16))) _Float16 v16h;
typedef __attribute__((ext_vector_type(8)))  float    v8f;
typedef __attribute__((ext_vector_type(4)))  unsigned int v4u;
typedef __attribute__((ext_vector_type(8)))  int      v8i;
typedef __attribute__((ext_vector_type(4)))  int      v4i;

// ---------------------------------------------------------------------------
// TDM: 2-D tile load (f16 elements) Global -> LDS via Tensor Data Mover.
// Descriptor per CDNA5 ISA (D# group0/group1).
// tdim* = remaining tensor extent from tile origin (OOB reads return zero).
// ---------------------------------------------------------------------------
__device__ __forceinline__ void tdm_load_tile_f16(
    unsigned lds_byte_off, const _Float16* tile_origin,
    unsigned tdim0, unsigned tdim1, unsigned stride0_elems,
    unsigned tile0, unsigned tile1)
{
    unsigned long long ga = (unsigned long long)(size_t)tile_origin;
    v4u g0;
    g0.x = 1u;                                        // count=1, user mode
    g0.y = lds_byte_off;                              // LDS byte address
    g0.z = (unsigned)(ga & 0xffffffffull);            // global_addr[31:0]
    g0.w = (unsigned)((ga >> 32) & 0x01ffffffull) | (2u << 30); // addr[56:32] | type=2
    v8i g1;
    g1[0] = (int)(1u << 16);                          // data_size=1 (2 bytes)
    g1[1] = (int)((tdim0 & 0xffffu) << 16);           // tensor_dim0[15:0] @bits63:48
    g1[2] = (int)((tdim0 >> 16) | ((tdim1 & 0xffffu) << 16)); // dim0[31:16], dim1[15:0]
    g1[3] = (int)((tdim1 >> 16) | (tile0 << 16));     // dim1[31:16], tile_dim0
    g1[4] = (int)(tile1 & 0xffffu);                   // tile_dim1 (tile_dim2=0)
    g1[5] = (int)stride0_elems;                       // tensor_dim0_stride[31:0]
    g1[6] = 0;                                        // stride0[47:32], dim1_stride lo
    g1[7] = 0;
    v4i z4 = {0, 0, 0, 0};
#if defined(__clang_major__) && (__clang_major__ >= 23)
    v8i z8 = {0, 0, 0, 0, 0, 0, 0, 0};
    __builtin_amdgcn_tensor_load_to_lds(g0, g1, z4, z4, z8, 0);
#else
    __builtin_amdgcn_tensor_load_to_lds(g0, g1, z4, z4, 0);
#endif
}

// ---------------------------------------------------------------------------
// WMMA GEMM with TDM staging: out = A @ W^T (+bias if F32OUT).
// A: [Nrows,K] f16, W: [Mout,K] f16 (both pre-converted).
// K % 32 == 0, Mout % 64 == 0. 128 threads / 4 waves; 64x64 tile per block.
// ---------------------------------------------------------------------------
template <bool F32OUT>
__global__ __launch_bounds__(128)
void gemm_wmma_tdm(const _Float16* __restrict__ A, const _Float16* __restrict__ W,
                   float* __restrict__ outF, _Float16* __restrict__ outH,
                   const float* __restrict__ bias,
                   int Nrows, int K, int Mout)
{
    __shared__ _Float16 As[64][32];
    __shared__ _Float16 Bs[64][32];

    const int tile_r = blockIdx.x * 64;
    const int tile_c = blockIdx.y * 64;
    const int tid  = threadIdx.x;
    const int wave = tid >> 5;
    const int lane = tid & 31;

    const unsigned ldsA = (unsigned)(size_t)(&As[0][0]);
    const unsigned ldsB = (unsigned)(size_t)(&Bs[0][0]);

    v8f acc[4] = {};

    for (int k0 = 0; k0 < K; k0 += 32) {
        if (wave == 0) {
            // A tile: rows tile_r..+63 (OOB rows -> zero via tensor_dim1)
            tdm_load_tile_f16(ldsA, A + (size_t)tile_r * K + k0,
                              (unsigned)(K - k0), (unsigned)(Nrows - tile_r),
                              (unsigned)K, 32u, 64u);
            // B tile: W rows tile_c..+63
            tdm_load_tile_f16(ldsB, W + (size_t)tile_c * K + k0,
                              (unsigned)(K - k0), (unsigned)(Mout - tile_c),
                              (unsigned)K, 32u, 64u);
            __builtin_amdgcn_s_wait_tensorcnt(0);
        }
        __syncthreads();

        // A fragment (16x32 f16): lanes 0-15: K 0..15; lanes 16-31: K 16..31
        v16h af;
        {
            int m  = (wave << 4) + (lane & 15);
            int kb = (lane >> 4) << 4;
            #pragma unroll
            for (int j = 0; j < 16; ++j) af[j] = As[m][kb + j];
        }
        #pragma unroll
        for (int s = 0; s < 4; ++s) {
            // B fragment (32x16 f16): col = lane&15; lanes 0-15 K 0..15, 16-31 K 16..31
            v16h bf;
            int n  = (s << 4) + (lane & 15);
            int kg = (lane >> 4) << 4;
            #pragma unroll
            for (int t = 0; t < 16; ++t) bf[t] = Bs[n][kg + t];
            acc[s] = __builtin_amdgcn_wmma_f32_16x16x32_f16(
                false, af, false, bf, (short)0, acc[s], false, false);
        }
        __syncthreads();
    }

    // C/D layout: VGPR j -> row ((lane>>4)*8 + j), col = lane&15
    const int rbase = tile_r + (wave << 4) + ((lane >> 4) << 3);
    const int cIn   = lane & 15;
    #pragma unroll
    for (int s = 0; s < 4; ++s) {
        int col = tile_c + (s << 4) + cIn;
        float bv = F32OUT ? bias[col] : 0.f;
        #pragma unroll
        for (int j = 0; j < 8; ++j) {
            int row = rbase + j;
            if (row < Nrows) {
                float v = acc[s][j] + bv;
                if (F32OUT) outF[(size_t)row * Mout + col] = v;
                else        outH[(size_t)row * Mout + col] = (_Float16)v;
            }
        }
    }
}

// ---------------------------------------------------------------------------
// helpers
// ---------------------------------------------------------------------------
__device__ inline void atomicMaxF(float* addr, float v) {
    if (v >= 0.f) atomicMax((int*)addr, __float_as_int(v));
    else          atomicMin((unsigned int*)addr, __float_as_uint(v));
}

__global__ void f32_to_f16(const float* __restrict__ src, _Float16* __restrict__ dst, int n) {
    int i = blockIdx.x * blockDim.x + threadIdx.x;
    if (i < n) dst[i] = (_Float16)src[i];
}

__global__ void fill_f32(float* p, float v, int n) {
    int i = blockIdx.x * blockDim.x + threadIdx.x;
    if (i < n) p[i] = v;
}

__global__ void init_out_bias(float* out, const float* __restrict__ bias, int Nn, int C) {
    int i = blockIdx.x * blockDim.x + threadIdx.x;
    if (i < Nn * C) out[i] = bias[i & (C - 1)];   // C is a power of two (128/256/512)
}

__global__ void relu_inplace(float* p, int n) {
    int i = blockIdx.x * blockDim.x + threadIdx.x;
    if (i < n) p[i] = fmaxf(p[i], 0.f);
}

// per-node attention scalars: as_out[i] = h[i].a_src, ad_out[i] = h[i].a_dst
__global__ void node_alpha(const _Float16* __restrict__ lin,
                           const float* __restrict__ a_src, const float* __restrict__ a_dst,
                           float* as_out, float* ad_out, int Nn, int C)
{
    int node = blockIdx.x * (blockDim.x >> 5) + (threadIdx.x >> 5);
    int lane = threadIdx.x & 31;
    if (node >= Nn) return;
    float s = 0.f, d = 0.f;
    for (int c = lane; c < C; c += 32) {
        float h = (float)lin[(size_t)node * C + c];
        s += h * a_src[c];
        d += h * a_dst[c];
    }
    #pragma unroll
    for (int off = 16; off; off >>= 1) {
        s += __shfl_xor(s, off);
        d += __shfl_xor(d, off);
    }
    if (lane == 0) { as_out[node] = s; ad_out[node] = d; }
}

// pass 1: leaky-relu logits + segment max over dst
__global__ void edge_logits(const int* __restrict__ ei, int E_, int Etot,
                            const float* __restrict__ asrc, const float* __restrict__ adst,
                            float* logit, float* m)
{
    int e = blockIdx.x * blockDim.x + threadIdx.x;
    if (e >= Etot) return;
    int s = (e < E_) ? ei[e]      : (e - E_);
    int d = (e < E_) ? ei[E_ + e] : (e - E_);
    float l = asrc[s] + adst[d];
    l = (l > 0.f) ? l : NEG_SLOPE * l;
    logit[e] = l;
    atomicMaxF(&m[d], l);
}

// pass 2: exp(logit - m[dst]) + segment sum
__global__ void edge_exp(const int* __restrict__ ei, int E_, int Etot,
                         const float* __restrict__ logit, const float* __restrict__ m,
                         float* eex, float* denom)
{
    int e = blockIdx.x * blockDim.x + threadIdx.x;
    if (e >= Etot) return;
    int d = (e < E_) ? ei[E_ + e] : (e - E_);
    float ex = __expf(logit[e] - m[d]);
    eex[e] = ex;
    atomicAdd(&denom[d], ex);
}

// pass 3: out[dst] += h[src] * alpha  (one wave per edge)
__global__ void edge_scatter(const int* __restrict__ ei, int E_, int Etot,
                             const _Float16* __restrict__ lin,
                             const float* __restrict__ eex, const float* __restrict__ denom,
                             float* out, int C)
{
    int e    = blockIdx.x * (blockDim.x >> 5) + (threadIdx.x >> 5);
    int lane = threadIdx.x & 31;
    if (e >= Etot) return;
    int s = (e < E_) ? ei[e]      : (e - E_);
    int d = (e < E_) ? ei[E_ + e] : (e - E_);
    float alpha = eex[e] / (denom[d] + 1e-16f);
    for (int c = lane; c < C; c += 32)
        atomicAdd(&out[(size_t)d * C + c], (float)lin[(size_t)s * C + c] * alpha);
}

// global max pool by batch id (one wave per node)
__global__ void pool_max(const float* __restrict__ h, const int* __restrict__ batch,
                         float* pooled, int Nn, int C)
{
    int node = blockIdx.x * (blockDim.x >> 5) + (threadIdx.x >> 5);
    int lane = threadIdx.x & 31;
    if (node >= Nn) return;
    int g = batch[node];
    for (int c = lane; c < C; c += 32)
        atomicMaxF(&pooled[(size_t)g * C + c], h[(size_t)node * C + c]);
}

// batchnorm over rows (population var), optional relu; one wave per column
__global__ void bn_cols(const float* __restrict__ z, const float* __restrict__ g,
                        const float* __restrict__ b, float* out,
                        int R, int Ccols, int doRelu)
{
    int col  = blockIdx.x * (blockDim.x >> 5) + (threadIdx.x >> 5);
    int lane = threadIdx.x & 31;
    if (col >= Ccols) return;
    float s = 0.f, s2 = 0.f;
    for (int r = lane; r < R; r += 32) {
        float v = z[(size_t)r * Ccols + col];
        s += v; s2 += v * v;
    }
    #pragma unroll
    for (int off = 16; off; off >>= 1) {
        s  += __shfl_xor(s, off);
        s2 += __shfl_xor(s2, off);
    }
    float mu  = s / (float)R;
    float var = s2 / (float)R - mu * mu;
    float inv = rsqrtf(var + BN_EPS) * g[col];
    float bb  = b[col];
    for (int r = lane; r < R; r += 32) {
        float v = z[(size_t)r * Ccols + col];
        float o = (v - mu) * inv + bb;
        if (doRelu) o = fmaxf(o, 0.f);
        out[(size_t)r * Ccols + col] = o;
    }
}

// ---------------------------------------------------------------------------
// launcher
// ---------------------------------------------------------------------------
extern "C" void kernel_launch(void* const* d_in, const int* in_sizes, int n_in,
                              void* d_out, int out_size, void* d_ws, size_t ws_size,
                              hipStream_t stream)
{
    (void)n_in; (void)ws_size;
    const float* x     = (const float*)d_in[0];
    const int*   ei    = (const int*)d_in[1];
    const int*   batch = (const int*)d_in[2];
    const float* W1  = (const float*)d_in[3];
    const float* as1 = (const float*)d_in[4];
    const float* ad1 = (const float*)d_in[5];
    const float* b1  = (const float*)d_in[6];
    const float* W2  = (const float*)d_in[7];
    const float* as2 = (const float*)d_in[8];
    const float* ad2 = (const float*)d_in[9];
    const float* b2  = (const float*)d_in[10];
    const float* W3  = (const float*)d_in[11];
    const float* as3 = (const float*)d_in[12];
    const float* ad3 = (const float*)d_in[13];
    const float* b3  = (const float*)d_in[14];
    const float* fc1_w = (const float*)d_in[15];
    const float* fc1_b = (const float*)d_in[16];
    const float* bn4_g = (const float*)d_in[17];
    const float* bn4_b = (const float*)d_in[18];
    const float* fc2_w = (const float*)d_in[19];
    const float* fc2_b = (const float*)d_in[20];
    const float* bn5_g = (const float*)d_in[21];
    const float* bn5_b = (const float*)d_in[22];

    const int N    = in_sizes[2];           // 100000
    const int E    = in_sizes[1] / 2;       // 1600000
    const int C    = in_sizes[4];           // 128
    const int Etot = E + N;                 // with self loops
    const int G    = out_size / (4 * C);    // 256

    // ---- workspace layout (floats) ----
    float* ws = (float*)d_ws;
    size_t off = 0;
    float* out3 = ws + off; off += (size_t)N * 4 * C;   // aliases out1 (first N*C, dead first)
    float* out1 = out3;
    float* out2 = ws + off; off += (size_t)N * 2 * C;
    _Float16* lin = (_Float16*)(ws + off); off += (size_t)N * 2 * C; // N*4C halves cap
    _Float16* Ah  = (_Float16*)(ws + off); off += (size_t)N * C;     // N*2C halves cap
    _Float16* Wh  = (_Float16*)(ws + off); off += 262144;            // 512K halves cap
    float* asrcA = ws + off; off += N;
    float* adstA = ws + off; off += N;
    float* mArr  = ws + off; off += N;
    float* denom = ws + off; off += N;
    float* elog  = ws + off; off += Etot;
    float* eex   = ws + off; off += Etot;
    float* pooled= ws + off; off += (size_t)G * 4 * C;
    float* z1    = ws + off; off += (size_t)G * 1024;
    float* z2    = ws + off; off += (size_t)G * 512;

    const int T = 256;
    auto cdiv = [](int a, int b) { return (a + b - 1) / b; };

    // GEMM front-end: convert f32 operands to f16 then run WMMA+TDM GEMM
    auto gemm_f16out = [&](const float* Xin, const float* Wm, _Float16* outH,
                           int Nrows, int K, int Mout) {
        f32_to_f16<<<cdiv(Nrows * K, T), T, 0, stream>>>(Xin, Ah, Nrows * K);
        f32_to_f16<<<cdiv(Mout * K, T), T, 0, stream>>>(Wm, Wh, Mout * K);
        dim3 gg(cdiv(Nrows, 64), Mout / 64);
        gemm_wmma_tdm<false><<<gg, 128, 0, stream>>>(Ah, Wh, nullptr, outH, nullptr,
                                                     Nrows, K, Mout);
    };
    auto gemm_f32out = [&](const float* Xin, const float* Wm, float* outF, const float* bi,
                           int Nrows, int K, int Mout) {
        f32_to_f16<<<cdiv(Nrows * K, T), T, 0, stream>>>(Xin, Ah, Nrows * K);
        f32_to_f16<<<cdiv(Mout * K, T), T, 0, stream>>>(Wm, Wh, Mout * K);
        dim3 gg(cdiv(Nrows, 64), Mout / 64);
        gemm_wmma_tdm<true><<<gg, 128, 0, stream>>>(Ah, Wh, outF, nullptr, bi,
                                                    Nrows, K, Mout);
    };

    // -------- one GAT layer --------
    auto gat_layer = [&](const float* Xin, int Cin, int Cout,
                         const float* W, const float* as_, const float* ad_,
                         const float* bi, float* outbuf) {
        gemm_f16out(Xin, W, lin, N, Cin, Cout);
        node_alpha<<<cdiv(N, 8), T, 0, stream>>>(lin, as_, ad_, asrcA, adstA, N, Cout);
        fill_f32<<<cdiv(N, T), T, 0, stream>>>(mArr, FLT_NEG_HUGE, N);
        fill_f32<<<cdiv(N, T), T, 0, stream>>>(denom, 0.f, N);
        init_out_bias<<<cdiv(N * Cout, T), T, 0, stream>>>(outbuf, bi, N, Cout);
        edge_logits<<<cdiv(Etot, T), T, 0, stream>>>(ei, E, Etot, asrcA, adstA, elog, mArr);
        edge_exp<<<cdiv(Etot, T), T, 0, stream>>>(ei, E, Etot, elog, mArr, eex, denom);
        edge_scatter<<<cdiv(Etot, 8), T, 0, stream>>>(ei, E, Etot, lin, eex, denom, outbuf, Cout);
        relu_inplace<<<cdiv(N * Cout, T), T, 0, stream>>>(outbuf, N * Cout);
    };

    gat_layer(x,    C,     C,     W1, as1, ad1, b1, out1);
    gat_layer(out1, C,     2 * C, W2, as2, ad2, b2, out2);
    gat_layer(out2, 2 * C, 4 * C, W3, as3, ad3, b3, out3);

    // -------- global max pool --------
    fill_f32<<<cdiv(G * 4 * C, T), T, 0, stream>>>(pooled, FLT_NEG_HUGE, G * 4 * C);
    pool_max<<<cdiv(N, 8), T, 0, stream>>>(out3, batch, pooled, N, 4 * C);

    // -------- MLP head --------
    gemm_f32out(pooled, fc1_w, z1, fc1_b, G, 4 * C, 1024);
    bn_cols<<<cdiv(1024, 8), T, 0, stream>>>(z1, bn4_g, bn4_b, z1, G, 1024, 1);
    gemm_f32out(z1, fc2_w, z2, fc2_b, G, 1024, 512);
    bn_cols<<<cdiv(512, 8), T, 0, stream>>>(z2, bn5_g, bn5_b, (float*)d_out, G, 512, 0);
}
